// ModulatedConv2d_69612829933579
// MI455X (gfx1250) — matrix operand order repaired and checked by reading, not compiled, not run on hardware
//
#include <hip/hip_runtime.h>

// ---------------------------------------------------------------------------
// StyleGAN2 modulated conv_transpose2d(stride=2) + blur for MI455X (gfx1250).
// Conv-transpose decomposed into 4 parity phases -> dense WMMA f16 GEMMs with
// fp32 accumulation (v_wmma_f32_16x16x32_f16). x is pre-converted to f16 with
// a zero halo so the LDS staging loop is branch-free u16->b32 traffic.
// ---------------------------------------------------------------------------

typedef __attribute__((ext_vector_type(16))) _Float16 v16h;
typedef __attribute__((ext_vector_type(8)))  float    v8f;

#define B_    16
#define CIN   256
#define COUT  128
#define SDIM  512
#define HW    64
#define W16   72           // padded f16 x row: [0]=zero halo, [1..64]=x, rest 0
#define HT    129          // conv-transpose output extent (2*64+1)
#define FO    128          // final output extent
#define SCALE (1.0f/48.0f) // 1/sqrt(Cin*K*K)
#define EPS_  1e-8f

// --------------------------- K0: style modulation ---------------------------
__global__ void k0_style(const float* __restrict__ style,
                         const float* __restrict__ mod_w,
                         const float* __restrict__ mod_b,
                         float* __restrict__ s) {
  int idx = blockIdx.x * blockDim.x + threadIdx.x;
  if (idx >= B_ * CIN) return;
  int b = idx >> 8, cin = idx & (CIN - 1);
  const float* st = style + b * SDIM;
  const float* mw = mod_w + cin * SDIM;
  float acc = mod_b[cin];
  for (int t = 0; t < SDIM; ++t) acc += st[t] * mw[t];
  s[idx] = acc;
}

// ------------------------------ K0b: demod ----------------------------------
__global__ void k0b_demod(const float* __restrict__ weight,
                          const float* __restrict__ s,
                          float* __restrict__ demod) {
  int idx = blockIdx.x * blockDim.x + threadIdx.x;
  if (idx >= B_ * COUT) return;
  int b = idx / COUT, co = idx % COUT;
  const float* sb = s + b * CIN;
  float sum = 0.f;
  for (int ci = 0; ci < CIN; ++ci) {
    float sv = SCALE * sb[ci];
    const float* wptr = weight + (size_t)(co * CIN + ci) * 9;
    for (int t = 0; t < 9; ++t) { float w = wptr[t] * sv; sum += w * w; }
  }
  demod[idx] = rsqrtf(sum + EPS_);
}

// ------------------------- K1: pack f16 A-matrices --------------------------
// Layout index: ((((b*9+tap)*8 + mt)*8 + kc)*32 + lane)*16 + i16
// 16x32 f16 A, contiguous k order used here (tileB reads match, not ISA table order): lane L -> m = L&15, h = L>>4;
//   element i16: k = h*16 + i16
__global__ void k1_pack(const float* __restrict__ weight,
                        const float* __restrict__ s,
                        const float* __restrict__ demod,
                        _Float16* __restrict__ wp) {
  int idx = blockIdx.x * blockDim.x + threadIdx.x;
  const int total = B_ * 9 * 8 * 8 * 32 * 16;  // 4,718,592
  if (idx >= total) return;
  int i16  = idx & 15;
  int lane = (idx >> 4) & 31;
  int kc   = (idx >> 9) & 7;
  int mt   = (idx >> 12) & 7;
  int rest = idx >> 15;          // b*9 + tap
  int tap  = rest % 9;
  int b    = rest / 9;
  int v = i16 >> 1, j = i16 & 1;
  int h = lane >> 4, m = lane & 15;
  int k    = (h << 4) + i16; (void)v; (void)j;
  int cin  = (kc << 5) + k;
  int cout = (mt << 4) + m;
  float val = SCALE * weight[(size_t)(cout * CIN + cin) * 9 + tap]
            * s[b * CIN + cin] * demod[b * COUT + cout];
  wp[idx] = (_Float16)val;
}

// ------------------ K1b: x -> f16 with zero-filled x-halo -------------------
// x16[b][cin][iy][hx], hx in [0,72): hx==0 and hx>=65 are zero, else x[hx-1].
__global__ void k1b_xhalf(const float* __restrict__ x,
                          unsigned short* __restrict__ x16) {
  int idx = blockIdx.x * blockDim.x + threadIdx.x;
  const int total = B_ * CIN * HW * W16;
  if (idx >= total) return;
  int hx   = idx % W16;
  int rest = idx / W16;          // (b*256+cin)*64 + iy
  _Float16 v = (_Float16)0.f;
  int ix = hx - 1;
  if ((unsigned)ix < (unsigned)HW) v = (_Float16)x[(size_t)rest * HW + ix];
  x16[idx] = __builtin_bit_cast(unsigned short, v);
}

// ----------------------- K2: phase GEMM (conv transpose) --------------------
// Block = 8 waves; wave mt owns cout tile [mt*16, mt*16+16) x 32 pixels of one
// output row oy=2a+py. Per K-chunk: stage a 2-row x 34-pixel(+-1 halo) x 32-cin
// f16 tile once, then up to 4 taps of WMMA read it at constant LDS offsets.
__global__ __launch_bounds__(256) void k2_convt(
    const unsigned short* __restrict__ x16,
    const _Float16* __restrict__ wp,
    float* __restrict__ cw, int py, int px) {
  // [r][p][cin], p in [0,34) covers ix = c0-1 .. c0+32; cin padded 32->40
  __shared__ __align__(16) unsigned short tileB[2 * 34 * 40];

  const int a    = blockIdx.y;
  const int c0   = blockIdx.x << 5;
  const int b    = blockIdx.z;
  const int tid  = threadIdx.x;
  const int lane = tid & 31;
  const int mt   = tid >> 5;
  const int h    = lane >> 4;
  const int nlo  = lane & 15;
  const int Nx   = (px == 0) ? 65 : 64;
  const int nky  = (py == 0) ? 2 : 1;
  const int nkx  = (px == 0) ? 2 : 1;

  v8f acc0 = {};
  v8f acc1 = {};

  const size_t xbase = (size_t)b * CIN * HW * W16;
  const int iy0c = min(max(a, 0), HW - 1);
  const int iy1c = min(max(a - 1, 0), HW - 1);

  for (int kc = 0; kc < 8; ++kc) {
    __syncthreads();
    // Stage 1088 dwords: item u = (cin-pair cp, row r, pixel p); halo buffer
    // means no bounds checks and no conversions -> straight-line u16 gathers.
    for (int u = tid; u < 1088; u += 256) {
      int cp = u / 68;                 // 0..15 (cin pair)
      int pl = u - cp * 68;            // 0..67
      int r  = (pl >= 34) ? 1 : 0;
      int p  = pl - r * 34;
      int iy = r ? iy1c : iy0c;
      int hx = c0 + p; if (hx > W16 - 1) hx = W16 - 1;
      size_t g = xbase + ((size_t)((kc << 5) + 2 * cp) * HW + iy) * W16 + hx;
      unsigned int lo = x16[g];
      unsigned int hi = x16[g + (size_t)HW * W16];
      ((unsigned int*)tileB)[(r * 34 + p) * 20 + cp] = lo | (hi << 16);
    }
    __syncthreads();

    for (int r = 0; r < nky; ++r) {
      if ((unsigned)(a - r) >= (unsigned)HW) continue;   // block-uniform
      for (int s = 0; s < nkx; ++s) {
        int tap = (py + 2 * r) * 3 + px + 2 * s;
        const _Float16* ap =
            wp + ((size_t)((b * 9 + tap) * 8 + mt) * 8 + kc) * 512 + lane * 16;
        const v16h aT = *(const v16h*)ap;
        __builtin_prefetch(ap + 512, 0, 3);              // next K-chunk A tile
        // tap x-shift is just an LDS base offset (rows are 80B => 16B aligned)
        const _Float16* bb =
            (const _Float16*)tileB + (r * 34 + 1 - s + nlo) * 40 + (h << 4);
        const v16h b0 = *(const v16h*)bb;
        const v16h b1 = *(const v16h*)(bb + 16 * 40);
        acc0 = __builtin_amdgcn_wmma_f32_16x16x32_f16(
            false, aT, false, b0, (short)0, acc0, false, false);
        acc1 = __builtin_amdgcn_wmma_f32_16x16x32_f16(
            false, aT, false, b1, (short)0, acc1, false, false);
      }
    }
  }

  // Store: C/D layout -> VGPR v holds M = v + 8*h, N = lane&15
  const int oy = 2 * a + py;
  size_t base0 = (((size_t)b * COUT + (mt << 4) + (h << 3)) * HT + oy) * HT;
  float* p0 = cw + base0 + 2 * (c0 + nlo) + px;          // subtile 0 column
  if (c0 + 32 <= Nx) {
    // Full tile (all interior blocks): unguarded, branch-free stores.
#pragma unroll
    for (int v = 0; v < 8; ++v) {
      p0[(size_t)v * (HT * HT)]      = acc0[v];
      p0[(size_t)v * (HT * HT) + 32] = acc1[v];          // +16 pixels * 2
    }
  } else {
    bool ok0 = (c0 + nlo) < Nx;
    bool ok1 = (c0 + nlo + 16) < Nx;
#pragma unroll
    for (int v = 0; v < 8; ++v) {
      if (ok0) p0[(size_t)v * (HT * HT)]      = acc0[v];
      if (ok1) p0[(size_t)v * (HT * HT) + 32] = acc1[v];
    }
  }
}

// ------------------------------- K3: blur -----------------------------------
// blurred[fy,fx] = sum_{u,v} cw[fy-1+u, fx-1+v] * k[u]*k[v]/16, k = {1,3,3,1}
__global__ void k3_blur(const float* __restrict__ cw, float* __restrict__ out) {
  int idx = blockIdx.x * blockDim.x + threadIdx.x;
  if (idx >= B_ * COUT * FO * FO) return;
  int fx = idx & 127;
  int fy = (idx >> 7) & 127;
  int ch = idx >> 14;
  const float* p = cw + (size_t)ch * HT * HT;
  const float kk[4] = {1.f, 3.f, 3.f, 1.f};
  float acc = 0.f;
  for (int u = 0; u < 4; ++u) {
    int oy = fy - 1 + u;
    if ((unsigned)oy >= (unsigned)HT) continue;
    float row = 0.f;
    for (int vv = 0; vv < 4; ++vv) {
      int ox = fx - 1 + vv;
      if ((unsigned)ox < (unsigned)HT) row += p[(size_t)oy * HT + ox] * kk[vv];
    }
    acc += row * kk[u];
  }
  out[idx] = acc * (1.0f / 16.0f);
}

// ------------------------------- launcher -----------------------------------
extern "C" void kernel_launch(void* const* d_in, const int* in_sizes, int n_in,
                              void* d_out, int out_size, void* d_ws, size_t ws_size,
                              hipStream_t stream) {
  const float* x      = (const float*)d_in[0];  // (16,256,64,64)
  const float* style  = (const float*)d_in[1];  // (16,512)
  const float* weight = (const float*)d_in[2];  // (1,128,256,3,3)
  const float* mod_w  = (const float*)d_in[3];  // (256,512)
  const float* mod_b  = (const float*)d_in[4];  // (256,)
  float* out = (float*)d_out;                    // (16,128,128,128)

  char* ws = (char*)d_ws;
  const size_t off_s     = 0;                               // 16 KB
  const size_t off_demod = 16384;                           // 8 KB
  const size_t off_wp    = 32768;                           // 9,437,184 B
  const size_t off_x16   = off_wp + (size_t)B_ * 9 * 8 * 8 * 32 * 16 * 2;
  const size_t off_cw    = off_x16 + (size_t)B_ * CIN * HW * W16 * 2;
  float*          s_buf  = (float*)(ws + off_s);
  float*          dm_buf = (float*)(ws + off_demod);
  _Float16*       wp_buf = (_Float16*)(ws + off_wp);
  unsigned short* x16buf = (unsigned short*)(ws + off_x16); // 37.7 MB
  float*          cw_buf = (float*)(ws + off_cw);           // 16*128*129*129 f32

  k0_style<<<dim3((B_ * CIN + 255) / 256), dim3(256), 0, stream>>>(
      style, mod_w, mod_b, s_buf);

  k0b_demod<<<dim3((B_ * COUT + 255) / 256), dim3(256), 0, stream>>>(
      weight, s_buf, dm_buf);

  {
    int total = B_ * 9 * 8 * 8 * 32 * 16;
    k1_pack<<<dim3((total + 255) / 256), dim3(256), 0, stream>>>(
        weight, s_buf, dm_buf, wp_buf);
  }

  {
    int total = B_ * CIN * HW * W16;
    k1b_xhalf<<<dim3((total + 255) / 256), dim3(256), 0, stream>>>(x, x16buf);
  }

  for (int py = 0; py < 2; ++py) {
    for (int px = 0; px < 2; ++px) {
      int Na = (py == 0) ? 65 : 64;
      int Nx = (px == 0) ? 65 : 64;
      dim3 grid((Nx + 31) / 32, Na, B_);
      k2_convt<<<grid, dim3(256), 0, stream>>>(x16buf, wp_buf, cw_buf, py, px);
    }
  }

  {
    int total = B_ * COUT * FO * FO;
    k3_blur<<<dim3((total + 255) / 256), dim3(256), 0, stream>>>(cw_buf, out);
  }
}
